// SelfAttention_14491219657281
// MI455X (gfx1250) — matrix operand order repaired and checked by reading, not compiled, not run on hardware
//
#include <hip/hip_runtime.h>

// ---------------------------------------------------------------------------
// Self-attention forward for MI455X (gfx1250), f16 WMMA with f32 accumulate.
//   k0: W fp32 [K][N] -> Wt f16 [N][K]  (one-time transpose/convert)
//   k1: qkv = x @ W_in + b_in   -> Q[b,h,s,dh], K[b,h,s,dh], Vt[b,h,dh,s] (f16)
//       (W tiles streamed via global_load_async_to_lds_b128, double-buffered)
//   k2: flash attention per (b,h), online softmax  -> O[b,s,h*dh] (f16)
//   k3: out = O @ W_out + b_out (fp32)
// ---------------------------------------------------------------------------

typedef __attribute__((ext_vector_type(16))) _Float16 v16h;
typedef __attribute__((ext_vector_type(8)))  _Float16 v8h;
typedef __attribute__((ext_vector_type(8)))  float    v8f;
typedef __attribute__((ext_vector_type(4)))  float    v4f;

#define WMMA_F16(a, b, c) \
  __builtin_amdgcn_wmma_f32_16x16x32_f16(false, (a), false, (b), (short)0, (c), false, false)

constexpr int BATCH = 2;
constexpr int SEQ   = 2048;
constexpr int HEADS = 16;
constexpr int DH    = 64;
constexpr int DEMB  = 1024;

// ---- WMMA fragment loaders (ISA 7.12.2 wave32 layouts) ---------------------
// A 16x32 f16: lane = M; elems 0..7 -> K = half*8+e, 8..15 -> K = 16+half*8+e.
__device__ __forceinline__ v16h load_a_f32(const float* __restrict__ row, int k0, int hf) {
  const v4f* p0 = (const v4f*)(row + k0 + hf * 8);
  const v4f* p1 = (const v4f*)(row + k0 + 16 + hf * 8);
  v4f a0 = p0[0], a1 = p0[1], b0 = p1[0], b1 = p1[1];
  v16h r;
#pragma unroll
  for (int i = 0; i < 4; ++i) {
    r[i]      = (_Float16)a0[i];
    r[4 + i]  = (_Float16)a1[i];
    r[8 + i]  = (_Float16)b0[i];
    r[12 + i] = (_Float16)b1[i];
  }
  return r;
}

__device__ __forceinline__ v16h load_a_h(const _Float16* __restrict__ row, int k0, int hf) {
  v8h lo = *(const v8h*)(row + k0 + hf * 8);
  v8h hi = *(const v8h*)(row + k0 + 16 + hf * 8);
  v16h r;
#pragma unroll
  for (int i = 0; i < 8; ++i) { r[i] = lo[i]; r[8 + i] = hi[i]; }
  return r;
}

// B 32x16 f16: lane = N; elems 0..7 -> K = half*8+e, 8..15 -> K = 16+half*8+e.
// `coln` = column n's K-contiguous storage.
__device__ __forceinline__ v16h load_b_h(const _Float16* __restrict__ coln, int k0, int hf) {
  v8h lo = *(const v8h*)(coln + k0 + hf * 8);
  v8h hi = *(const v8h*)(coln + k0 + 16 + hf * 8);
  v16h r;
#pragma unroll
  for (int i = 0; i < 8; ++i) { r[i] = lo[i]; r[8 + i] = hi[i]; }
  return r;
}

// ---- CDNA5 async copy: global -> LDS, 16 bytes per lane (ASYNCcnt) ---------
__device__ __forceinline__ void async_copy_b128(const void* g, void* lds) {
  // LDS aperture truncates flat addresses to addr[31:0] (ISA 10.2), so the low
  // 32 bits of the flat shared-pointer are the LDS byte address.
  unsigned lds_addr = (unsigned)(uintptr_t)lds;
  asm volatile("global_load_async_to_lds_b128 %0, %1, off"
               :: "v"(lds_addr), "v"((unsigned long long)(uintptr_t)g)
               : "memory");
}

// ---------------------------------------------------------------------------
// k0: Wt[n][k] f16  <-  W[k][n] fp32   (32x32 LDS tile transpose)
// ---------------------------------------------------------------------------
__global__ __launch_bounds__(256) void convert_w_f16t(
    const float* __restrict__ W, _Float16* __restrict__ Wt, int Kdim, int Ncols) {
  __shared__ float tile[32][33];
  const int n0 = blockIdx.x * 32, k0 = blockIdx.y * 32;
  const int tx = threadIdx.x & 31, ty = threadIdx.x >> 5;
#pragma unroll
  for (int i = 0; i < 32; i += 8)
    tile[ty + i][tx] = W[(size_t)(k0 + ty + i) * Ncols + n0 + tx];
  __syncthreads();
#pragma unroll
  for (int i = 0; i < 32; i += 8)
    Wt[(size_t)(n0 + ty + i) * Kdim + k0 + tx] = (_Float16)tile[tx][ty + i];
}

// ---------------------------------------------------------------------------
// GEMM: C[M x N] = A[M x K] * Wt^T + bias, Wt pre-transposed f16 [N][K].
// Block = 128x128 C tile, 8 waves (2 M x 4 N), each wave 64x32, k-step 32.
// Wt tiles double-buffered in LDS via async-to-LDS.
// MODE 0: A = fp32, epilogue scatters f16 Q/K/Vt.  MODE 1: A = f16, fp32 out.
// ---------------------------------------------------------------------------
constexpr int WT_PITCH = 40;  // halves; conflict-free across 16 lanes

template <int MODE>
__global__ __launch_bounds__(256) void gemm_f16wmma(
    const float* __restrict__ Af, const _Float16* __restrict__ Ah,
    const _Float16* __restrict__ Wt, const float* __restrict__ bias,
    _Float16* __restrict__ Qo, _Float16* __restrict__ Ko, _Float16* __restrict__ Vto,
    float* __restrict__ Fo, int Kdim, int Ncols) {
  __shared__ _Float16 WtBuf[2][128 * WT_PITCH];  // 2 x 10 KB

  const int tid   = threadIdx.x;
  const int lane  = tid & 31;
  const int wid   = tid >> 5;
  const int ln    = lane & 15;
  const int hf    = lane >> 4;
  const int waveM = wid >> 2;
  const int waveN = wid & 3;
  const int mBase  = blockIdx.y * 128 + waveM * 64;
  const int nBase0 = blockIdx.x * 128;
  const int nBaseW = nBase0 + waveN * 32;

  v8f acc[4][2];
#pragma unroll
  for (int mt = 0; mt < 4; ++mt)
#pragma unroll
    for (int nt = 0; nt < 2; ++nt)
#pragma unroll
      for (int e = 0; e < 8; ++e) acc[mt][nt][e] = 0.f;

  // stage Wt[nBase0..+127][k0..k0+31] into WtBuf[buf]: 512 16B chunks,
  // 2 per thread (chunk = 8 halves of k for one n-row)
  auto stage = [&](int k0, int buf) {
#pragma unroll
    for (int j = 0; j < 2; ++j) {
      int c  = j * 256 + tid;
      int n  = c >> 2;
      int kc = (c & 3) * 8;
      async_copy_b128(Wt + (size_t)(nBase0 + n) * Kdim + k0 + kc,
                      &WtBuf[buf][n * WT_PITCH + kc]);
    }
  };

  stage(0, 0);
  for (int k0 = 0; k0 < Kdim; k0 += 32) {
    const int buf = (k0 >> 5) & 1;
    if (k0 + 32 < Kdim) {
      stage(k0 + 32, buf ^ 1);
      asm volatile("s_wait_asynccnt 0x2" ::: "memory");  // current buf resident
    } else {
      asm volatile("s_wait_asynccnt 0x0" ::: "memory");
    }
    __syncthreads();

    v16h af[4];
#pragma unroll
    for (int mt = 0; mt < 4; ++mt) {
      int row = mBase + mt * 16 + ln;
      if (MODE == 0) af[mt] = load_a_f32(Af + (size_t)row * Kdim, k0, hf);
      else           af[mt] = load_a_h(Ah + (size_t)row * Kdim, k0, hf);
    }
#pragma unroll
    for (int nt = 0; nt < 2; ++nt) {
      v16h bf = load_b_h(&WtBuf[buf][(waveN * 32 + nt * 16 + ln) * WT_PITCH], 0, hf);
#pragma unroll
      for (int mt = 0; mt < 4; ++mt) acc[mt][nt] = WMMA_F16(af[mt], bf, acc[mt][nt]);
    }
    __syncthreads();
  }

  // epilogue (C layout: lane -> col, elem r -> row r + 8*half)
#pragma unroll
  for (int nt = 0; nt < 2; ++nt) {
    int col  = nBaseW + nt * 16 + ln;
    float bv = bias[col];
#pragma unroll
    for (int mt = 0; mt < 4; ++mt) {
#pragma unroll
      for (int e = 0; e < 8; ++e) {
        int row = mBase + mt * 16 + hf * 8 + e;
        float v = acc[mt][nt][e] + bv;
        if (MODE == 1) {
          Fo[(size_t)row * Ncols + col] = v;
        } else {
          int b = row >> 11, s = row & (SEQ - 1);
          int seg = col >> 10;            // 0=Q, 1=K, 2=V
          int cc = col & (DEMB - 1);
          int h = cc >> 6, dh = cc & (DH - 1);
          size_t ho = (size_t)(b * HEADS + h);
          if (seg == 0)      Qo[(ho * SEQ + s) * DH + dh] = (_Float16)v;
          else if (seg == 1) Ko[(ho * SEQ + s) * DH + dh] = (_Float16)v;
          else               Vto[(ho * DH + dh) * SEQ + s] = (_Float16)v;  // V^T
        }
      }
    }
  }
}

// ---------------------------------------------------------------------------
// Flash attention: block = (b, h, 128-query tile), 4 waves x 32 q-rows.
// Per 32-key tile: S = Q*K^T (8 WMMA), online softmax, O += P*V (8 WMMA).
// ---------------------------------------------------------------------------
constexpr int P_PITCH = 40;

__global__ __launch_bounds__(128) void attn_f16wmma(
    const _Float16* __restrict__ Q, const _Float16* __restrict__ K,
    const _Float16* __restrict__ Vt, _Float16* __restrict__ O) {
  __shared__ _Float16 Plds[4 * 32 * P_PITCH];  // per-wave 32x32 P tile

  const int tid  = threadIdx.x;
  const int lane = tid & 31;
  const int w    = tid >> 5;
  const int ln   = lane & 15;
  const int hf   = lane >> 4;
  const int h    = blockIdx.y;
  const int b    = blockIdx.z;
  const int qBase = blockIdx.x * 128 + w * 32;

  const size_t hoQK = (size_t)(b * HEADS + h) * SEQ * DH;
  const _Float16* Qh  = Q + hoQK;
  const _Float16* Kh  = K + hoQK;
  const _Float16* Vth = Vt + (size_t)(b * HEADS + h) * DH * SEQ;
  _Float16* Pw = &Plds[w * 32 * P_PITCH];

  // persistent Q fragments: 2 m-tiles x 2 k-steps over Dh=64
  v16h qf[2][2];
#pragma unroll
  for (int mt = 0; mt < 2; ++mt) {
    const _Float16* rp = Qh + (size_t)(qBase + mt * 16 + ln) * DH;
#pragma unroll
    for (int ks = 0; ks < 2; ++ks) qf[mt][ks] = load_a_h(rp, ks * 32, hf);
  }

  v8f o[2][4], m[2], lsum[2];
#pragma unroll
  for (int mt = 0; mt < 2; ++mt)
#pragma unroll
    for (int e = 0; e < 8; ++e) {
      m[mt][e] = -3.0e38f;
      lsum[mt][e] = 0.f;
#pragma unroll
      for (int nt = 0; nt < 4; ++nt) o[mt][nt][e] = 0.f;
    }

  const float sc = 0.125f * 1.44269504f;  // 1/sqrt(Dh) * log2(e)

  for (int kt = 0; kt < SEQ / 32; ++kt) {
    const int keyBase = kt * 32;

    // scores S = Q * K^T  (B column n == key row of K, contiguous over dh)
    v8f s[2][2];
#pragma unroll
    for (int mt = 0; mt < 2; ++mt)
#pragma unroll
      for (int nt = 0; nt < 2; ++nt)
#pragma unroll
        for (int e = 0; e < 8; ++e) s[mt][nt][e] = 0.f;

#pragma unroll
    for (int ks = 0; ks < 2; ++ks) {
      v16h kf[2];
#pragma unroll
      for (int nt = 0; nt < 2; ++nt)
        kf[nt] = load_b_h(Kh + (size_t)(keyBase + nt * 16 + ln) * DH, ks * 32, hf);
#pragma unroll
      for (int mt = 0; mt < 2; ++mt)
#pragma unroll
        for (int nt = 0; nt < 2; ++nt) s[mt][nt] = WMMA_F16(qf[mt][ks], kf[nt], s[mt][nt]);
    }

    // online softmax; row stats share the (elem r -> row r+8*half) layout
#pragma unroll
    for (int mt = 0; mt < 2; ++mt) {
      v8f t;
#pragma unroll
      for (int e = 0; e < 8; ++e) {
        float a0 = s[mt][0][e] * sc, a1 = s[mt][1][e] * sc;
        s[mt][0][e] = a0; s[mt][1][e] = a1;
        t[e] = fmaxf(a0, a1);
      }
#pragma unroll
      for (int mask = 1; mask <= 8; mask <<= 1)
#pragma unroll
        for (int e = 0; e < 8; ++e) t[e] = fmaxf(t[e], __shfl_xor(t[e], mask));

      v8f alpha, rs;
#pragma unroll
      for (int e = 0; e < 8; ++e) {
        float mn = fmaxf(m[mt][e], t[e]);
        alpha[e] = exp2f(m[mt][e] - mn);
        m[mt][e] = mn;
        float p0 = exp2f(s[mt][0][e] - mn);
        float p1 = exp2f(s[mt][1][e] - mn);
        s[mt][0][e] = p0; s[mt][1][e] = p1;
        rs[e] = p0 + p1;
      }
#pragma unroll
      for (int mask = 1; mask <= 8; mask <<= 1)
#pragma unroll
        for (int e = 0; e < 8; ++e) rs[e] += __shfl_xor(rs[e], mask);
#pragma unroll
      for (int e = 0; e < 8; ++e) lsum[mt][e] = lsum[mt][e] * alpha[e] + rs[e];
#pragma unroll
      for (int nt = 0; nt < 4; ++nt)
#pragma unroll
        for (int e = 0; e < 8; ++e) o[mt][nt][e] *= alpha[e];

      // P tile -> per-wave LDS (f16), C layout -> [row][key]
#pragma unroll
      for (int nt = 0; nt < 2; ++nt)
#pragma unroll
        for (int e = 0; e < 8; ++e)
          Pw[(mt * 16 + hf * 8 + e) * P_PITCH + nt * 16 + ln] = (_Float16)s[mt][nt][e];
    }

    // O += P * V   (A = P from LDS; B column n == dh row of Vt, contiguous keys)
    v16h pf[2];
#pragma unroll
    for (int mt = 0; mt < 2; ++mt) pf[mt] = load_a_h(Pw + (mt * 16 + ln) * P_PITCH, 0, hf);
#pragma unroll
    for (int nt = 0; nt < 4; ++nt) {
      v16h vf = load_b_h(Vth + (size_t)(nt * 16 + ln) * SEQ + keyBase, 0, hf);
#pragma unroll
      for (int mt = 0; mt < 2; ++mt) o[mt][nt] = WMMA_F16(pf[mt], vf, o[mt][nt]);
    }
  }

  // normalize and write O[b, s, h*64+dh] as f16
#pragma unroll
  for (int mt = 0; mt < 2; ++mt) {
    v8f inv;
#pragma unroll
    for (int e = 0; e < 8; ++e) inv[e] = 1.0f / lsum[mt][e];
#pragma unroll
    for (int nt = 0; nt < 4; ++nt) {
      int col = h * DH + nt * 16 + ln;
#pragma unroll
      for (int e = 0; e < 8; ++e) {
        int srow = qBase + mt * 16 + hf * 8 + e;
        O[(size_t)(b * SEQ + srow) * DEMB + col] = (_Float16)(o[mt][nt][e] * inv[e]);
      }
    }
  }
}

// ---------------------------------------------------------------------------
extern "C" void kernel_launch(void* const* d_in, const int* in_sizes, int n_in,
                              void* d_out, int out_size, void* d_ws, size_t ws_size,
                              hipStream_t stream) {
  const float* x     = (const float*)d_in[0];
  const float* W_in  = (const float*)d_in[1];
  const float* b_in  = (const float*)d_in[2];
  const float* W_out = (const float*)d_in[3];
  const float* b_out = (const float*)d_in[4];
  float* out = (float*)d_out;

  const size_t headElems = (size_t)BATCH * HEADS * SEQ * DH;  // 4 Mi f16 each
  _Float16* Qb    = (_Float16*)d_ws;
  _Float16* Kb    = Qb + headElems;
  _Float16* Vtb   = Kb + headElems;
  _Float16* Ob    = Vtb + headElems;
  _Float16* WtIn  = Ob + headElems;                 // [3D][D] f16
  _Float16* WtOut = WtIn + (size_t)3 * DEMB * DEMB; // [D][D]  f16
  // total workspace: 40 MB

  // one-time weight transpose/convert
  convert_w_f16t<<<dim3((3 * DEMB) / 32, DEMB / 32), 256, 0, stream>>>(
      W_in, WtIn, DEMB, 3 * DEMB);
  convert_w_f16t<<<dim3(DEMB / 32, DEMB / 32), 256, 0, stream>>>(
      W_out, WtOut, DEMB, DEMB);

  dim3 g1((3 * DEMB) / 128, (BATCH * SEQ) / 128);  // 24 x 32
  gemm_f16wmma<0><<<g1, 256, 0, stream>>>(x, nullptr, WtIn, b_in, Qb, Kb, Vtb,
                                          nullptr, DEMB, 3 * DEMB);

  dim3 g2(SEQ / 128, HEADS, BATCH);                // 16 x 16 x 2
  attn_f16wmma<<<g2, 128, 0, stream>>>(Qb, Kb, Vtb, Ob);

  dim3 g3(DEMB / 128, (BATCH * SEQ) / 128);        // 8 x 32
  gemm_f16wmma<1><<<g3, 256, 0, stream>>>(nullptr, Ob, WtOut, b_out, nullptr,
                                          nullptr, nullptr, out, DEMB, DEMB);
}
